// EncoderDecoderLSTM_7773890805989
// MI455X (gfx1250) — compile-verified, matrix operand order checked by reading, not run on hardware
//
#include <hip/hip_runtime.h>

#define Bdim 256
#define Tdim 512
#define Hdim 512
#define NPREDS 128
#define G4H 2048   // 4*H gate width

typedef __attribute__((ext_vector_type(16))) _Float16 v16h;
typedef __attribute__((ext_vector_type(8)))  _Float16 v8h;
typedef __attribute__((ext_vector_type(8)))  float    v8f;

__device__ __forceinline__ float sigmf(float x) { return 1.0f / (1.0f + __expf(-x)); }

__device__ __forceinline__ v8f wmma16(v16h a, v16h b, v8f c) {
  // v_wmma_f32_16x16x32_f16, f32 accumulate
  return __builtin_amdgcn_wmma_f32_16x16x32_f16(false, a, false, b, (short)0, c, false, false);
}

// A-fragment (16x32 f16, ISA 7.12.2): lane<16 -> M=lane, K = {0..7, 16..23};
// lane>=16 -> M=lane-16, K = {8..15, 24..31}. Two contiguous 16B loads per lane.
__device__ __forceinline__ v16h load_a_frag(const _Float16* __restrict__ A, int row, int kt, int lane) {
  const _Float16* p = A + (size_t)row * Hdim + kt * 32 + ((lane & 16) ? 8 : 0);
  v8h lo = *(const v8h*)p;
  v8h hi = *(const v8h*)(p + 16);
  v16h r;
#pragma unroll
  for (int j = 0; j < 8; ++j) { r[j] = lo[j]; r[8 + j] = hi[j]; }
  return r;
}

// ---------------- weight pre-pack (fp32 NxK row-major -> fp16 B-fragment-major) ---
// Fragment f=(nt*KT+kt) holds 512 halfs: lane*16+j  ->  W[n=nt*16+(lane&15),
// k=kt*32 + (lane&16?8:0) + j + (j>=8?8:0)]  (mirror of A layout, N striped on lanes)
__global__ __launch_bounds__(256) void pack_w_kernel(const float* __restrict__ W,
                                                     _Float16* __restrict__ out,
                                                     int N, int K) {
  int idx = blockIdx.x * 256 + threadIdx.x;
  if (idx >= N * K) return;
  int f = idx >> 9, rl = idx & 511;
  int lanep = rl >> 4, j = rl & 15;
  int KT = K >> 5;
  int nt = f / KT, kt = f - nt * KT;
  int n = nt * 16 + (lanep & 15);
  int k = kt * 32 + ((lanep & 16) ? 8 : 0) + j + ((j >= 8) ? 8 : 0);
  out[idx] = (_Float16)W[(size_t)n * K + k];
}

// Packed concat [W0 ; W1] along K (each N x Khalf), K = 2*Khalf
__global__ __launch_bounds__(256) void pack_wcat_kernel(const float* __restrict__ W0,
                                                        const float* __restrict__ W1,
                                                        _Float16* __restrict__ out,
                                                        int N, int Khalf) {
  int K = Khalf * 2;
  int idx = blockIdx.x * 256 + threadIdx.x;
  if (idx >= N * K) return;
  int f = idx >> 9, rl = idx & 511;
  int lanep = rl >> 4, j = rl & 15;
  int KT = K >> 5;
  int nt = f / KT, kt = f - nt * KT;
  int n = nt * 16 + (lanep & 15);
  int k = kt * 32 + ((lanep & 16) ? 8 : 0) + j + ((j >= 8) ? 8 : 0);
  float v = (k < Khalf) ? W0[(size_t)n * Khalf + k] : W1[(size_t)n * Khalf + (k - Khalf)];
  out[idx] = (_Float16)v;
}

__global__ __launch_bounds__(256) void zero_state_kernel(float* c0, float* c1,
                                                         _Float16* h0, _Float16* h1) {
  int i = blockIdx.x * 256 + threadIdx.x;
  if (i < Bdim * Hdim) {
    c0[i] = 0.0f; c1[i] = 0.0f;
    h0[i] = (_Float16)0.0f; h1[i] = (_Float16)0.0f;
  }
}

// ---------------- fused LSTM cell, layer-0 style (scalar input projection) -------
// gates = x*Wih + b + h_in @ Whh^T ; block = (batch tile 16) x (hidden chunk 64),
// wave w (0..3) computes gate w. Then fused pointwise cell update.
__global__ __launch_bounds__(128) void cell_l0_kernel(
    const _Float16* __restrict__ Wp,    // packed 2048 x 512
    const float* __restrict__ bias,     // 2048
    const float* __restrict__ Wih,      // 2048 (input dim 1)
    const float* __restrict__ xt, int xstride,
    const _Float16* __restrict__ h_in,  // B x H fp16
    _Float16* __restrict__ h_out,       // B x H fp16
    float* __restrict__ c)              // B x H fp32, in-place
{
  __shared__ float gl[4][16][64];
  const int mt = blockIdx.x, hc = blockIdx.y;
  const int wave = threadIdx.x >> 5, lane = threadIdx.x & 31;
  const int KT = Hdim / 32;                 // 16
  const int row = mt * 16 + (lane & 15);
  const int nt0 = wave * (Hdim / 16) + hc * 4;   // fragment N-tile base

  v8f a0 = {}, a1 = {}, a2 = {}, a3 = {};
  for (int kt = 0; kt < KT; ++kt) {
    v16h af = load_a_frag(h_in, row, kt, lane);
    v16h b0 = ((const v16h*)(Wp + ((size_t)(nt0 + 0) * KT + kt) * 512))[lane];
    v16h b1 = ((const v16h*)(Wp + ((size_t)(nt0 + 1) * KT + kt) * 512))[lane];
    v16h b2 = ((const v16h*)(Wp + ((size_t)(nt0 + 2) * KT + kt) * 512))[lane];
    v16h b3 = ((const v16h*)(Wp + ((size_t)(nt0 + 3) * KT + kt) * 512))[lane];
    a0 = wmma16(af, b0, a0);
    a1 = wmma16(af, b1, a1);
    a2 = wmma16(af, b2, a2);
    a3 = wmma16(af, b3, a3);
  }
  // C/D layout: VGPR r: lane<16 -> (M=r, N=lane); lane>=16 -> (M=r+8, N=lane-16)
  {
    int cn = lane & 15, ro = (lane >> 4) << 3;
#pragma unroll
    for (int r = 0; r < 8; ++r) {
      gl[wave][ro + r][cn]      = a0[r];
      gl[wave][ro + r][16 + cn] = a1[r];
      gl[wave][ro + r][32 + cn] = a2[r];
      gl[wave][ro + r][48 + cn] = a3[r];
    }
  }
  __syncthreads();

  const int r  = threadIdx.x >> 3;        // 0..15 batch row in tile
  const int cb = (threadIdx.x & 7) * 8;   // col base in chunk
  const int b  = mt * 16 + r;
  const float xv = xt[(size_t)b * xstride];
#pragma unroll
  for (int e = 0; e < 8; ++e) {
    int hid = hc * 64 + cb + e;
    float gi = gl[0][r][cb + e] + bias[hid]            + xv * Wih[hid];
    float gf = gl[1][r][cb + e] + bias[Hdim + hid]     + xv * Wih[Hdim + hid];
    float gg = gl[2][r][cb + e] + bias[2 * Hdim + hid] + xv * Wih[2 * Hdim + hid];
    float go = gl[3][r][cb + e] + bias[3 * Hdim + hid] + xv * Wih[3 * Hdim + hid];
    float cp = c[(size_t)b * Hdim + hid];
    float cn2 = sigmf(gf) * cp + sigmf(gi) * tanhf(gg);
    float hn  = sigmf(go) * tanhf(cn2);
    c[(size_t)b * Hdim + hid]    = cn2;
    h_out[(size_t)b * Hdim + hid] = (_Float16)hn;
  }
}

// ---------------- fused LSTM cell, layer-1 style ---------------------------------
// gates = b + [h_lower ; h_in] @ [Wih ; Whh]^T  (K = 1024, packed concat weights)
__global__ __launch_bounds__(128) void cell_l1_kernel(
    const _Float16* __restrict__ Wp,      // packed 2048 x 1024
    const float* __restrict__ bias,       // 2048
    const _Float16* __restrict__ h_low,   // B x H fp16 (this step's lower-layer h)
    const _Float16* __restrict__ h_in,    // B x H fp16 (this layer, prev step)
    _Float16* __restrict__ h_out,
    float* __restrict__ c)
{
  __shared__ float gl[4][16][64];
  const int mt = blockIdx.x, hc = blockIdx.y;
  const int wave = threadIdx.x >> 5, lane = threadIdx.x & 31;
  const int KT = 1024 / 32;               // 32 fragments along K
  const int row = mt * 16 + (lane & 15);
  const int nt0 = wave * (Hdim / 16) + hc * 4;

  v8f a0 = {}, a1 = {}, a2 = {}, a3 = {};
  for (int kt = 0; kt < 16; ++kt) {       // K 0..511 : h_low
    v16h af = load_a_frag(h_low, row, kt, lane);
    v16h b0 = ((const v16h*)(Wp + ((size_t)(nt0 + 0) * KT + kt) * 512))[lane];
    v16h b1 = ((const v16h*)(Wp + ((size_t)(nt0 + 1) * KT + kt) * 512))[lane];
    v16h b2 = ((const v16h*)(Wp + ((size_t)(nt0 + 2) * KT + kt) * 512))[lane];
    v16h b3 = ((const v16h*)(Wp + ((size_t)(nt0 + 3) * KT + kt) * 512))[lane];
    a0 = wmma16(af, b0, a0); a1 = wmma16(af, b1, a1);
    a2 = wmma16(af, b2, a2); a3 = wmma16(af, b3, a3);
  }
  for (int kt = 0; kt < 16; ++kt) {       // K 512..1023 : h_in (recurrent)
    v16h af = load_a_frag(h_in, row, kt, lane);
    int ktg = 16 + kt;
    v16h b0 = ((const v16h*)(Wp + ((size_t)(nt0 + 0) * KT + ktg) * 512))[lane];
    v16h b1 = ((const v16h*)(Wp + ((size_t)(nt0 + 1) * KT + ktg) * 512))[lane];
    v16h b2 = ((const v16h*)(Wp + ((size_t)(nt0 + 2) * KT + ktg) * 512))[lane];
    v16h b3 = ((const v16h*)(Wp + ((size_t)(nt0 + 3) * KT + ktg) * 512))[lane];
    a0 = wmma16(af, b0, a0); a1 = wmma16(af, b1, a1);
    a2 = wmma16(af, b2, a2); a3 = wmma16(af, b3, a3);
  }
  {
    int cn = lane & 15, ro = (lane >> 4) << 3;
#pragma unroll
    for (int r = 0; r < 8; ++r) {
      gl[wave][ro + r][cn]      = a0[r];
      gl[wave][ro + r][16 + cn] = a1[r];
      gl[wave][ro + r][32 + cn] = a2[r];
      gl[wave][ro + r][48 + cn] = a3[r];
    }
  }
  __syncthreads();

  const int r  = threadIdx.x >> 3;
  const int cb = (threadIdx.x & 7) * 8;
  const int b  = mt * 16 + r;
#pragma unroll
  for (int e = 0; e < 8; ++e) {
    int hid = hc * 64 + cb + e;
    float gi = gl[0][r][cb + e] + bias[hid];
    float gf = gl[1][r][cb + e] + bias[Hdim + hid];
    float gg = gl[2][r][cb + e] + bias[2 * Hdim + hid];
    float go = gl[3][r][cb + e] + bias[3 * Hdim + hid];
    float cp = c[(size_t)b * Hdim + hid];
    float cn2 = sigmf(gf) * cp + sigmf(gi) * tanhf(gg);
    float hn  = sigmf(go) * tanhf(cn2);
    c[(size_t)b * Hdim + hid]    = cn2;
    h_out[(size_t)b * Hdim + hid] = (_Float16)hn;
  }
}

// ---------------- decoder linear head: pred[b] = h1[b,:] . linW + linb -----------
__global__ __launch_bounds__(128) void pred_kernel(const _Float16* __restrict__ h1,
                                                   const float* __restrict__ linW,
                                                   const float* __restrict__ linb,
                                                   float* __restrict__ outcol, // d_out + s
                                                   float* __restrict__ prev) {
  __shared__ float red[128];
  int b = blockIdx.x;
  float s = 0.0f;
  for (int k = threadIdx.x; k < Hdim; k += 128)
    s += (float)h1[(size_t)b * Hdim + k] * linW[k];
  red[threadIdx.x] = s;
  __syncthreads();
  for (int off = 64; off > 0; off >>= 1) {
    if (threadIdx.x < off) red[threadIdx.x] += red[threadIdx.x + off];
    __syncthreads();
  }
  if (threadIdx.x == 0) {
    float p = red[0] + linb[0];
    outcol[(size_t)b * NPREDS] = p;
    prev[b] = p;
  }
}

extern "C" void kernel_launch(void* const* d_in, const int* in_sizes, int n_in,
                              void* d_out, int out_size, void* d_ws, size_t ws_size,
                              hipStream_t stream) {
  const float* x        = (const float*)d_in[0];
  const float* enc_Wih0 = (const float*)d_in[1];
  const float* enc_Whh0 = (const float*)d_in[2];
  const float* enc_b0   = (const float*)d_in[3];
  const float* enc_Wih1 = (const float*)d_in[4];
  const float* enc_Whh1 = (const float*)d_in[5];
  const float* enc_b1   = (const float*)d_in[6];
  const float* dec_Wih0 = (const float*)d_in[7];
  const float* dec_Whh0 = (const float*)d_in[8];
  const float* dec_b0   = (const float*)d_in[9];
  const float* dec_Wih1 = (const float*)d_in[10];
  const float* dec_Whh1 = (const float*)d_in[11];
  const float* dec_b1   = (const float*)d_in[12];
  const float* lin_W    = (const float*)d_in[13];
  const float* lin_b    = (const float*)d_in[14];
  float* out = (float*)d_out;

  const size_t MB = 1u << 20, KB = 1u << 10;
  char* ws = (char*)d_ws;
  _Float16* Wp_e0 = (_Float16*)(ws + 0);        // 2 MB  (2048x512 f16)
  _Float16* Wp_e1 = (_Float16*)(ws + 2 * MB);   // 4 MB  (2048x1024 f16)
  _Float16* Wp_d0 = (_Float16*)(ws + 6 * MB);   // 2 MB
  _Float16* Wp_d1 = (_Float16*)(ws + 8 * MB);   // 4 MB
  _Float16* h0buf[2] = { (_Float16*)(ws + 12 * MB),
                         (_Float16*)(ws + 12 * MB + 256 * KB) };
  _Float16* h1buf[2] = { (_Float16*)(ws + 12 * MB + 512 * KB),
                         (_Float16*)(ws + 12 * MB + 768 * KB) };
  float* c0   = (float*)(ws + 13 * MB);          // 512 KB
  float* c1   = (float*)(ws + 13 * MB + 512 * KB);
  float* prev = (float*)(ws + 14 * MB);          // B floats

  // one-time (per call) weight pack + state zero-init
  pack_w_kernel  <<<(G4H * Hdim) / 256, 256, 0, stream>>>(enc_Whh0, Wp_e0, G4H, Hdim);
  pack_wcat_kernel<<<(G4H * 2 * Hdim) / 256, 256, 0, stream>>>(enc_Wih1, enc_Whh1, Wp_e1, G4H, Hdim);
  pack_w_kernel  <<<(G4H * Hdim) / 256, 256, 0, stream>>>(dec_Whh0, Wp_d0, G4H, Hdim);
  pack_wcat_kernel<<<(G4H * 2 * Hdim) / 256, 256, 0, stream>>>(dec_Wih1, dec_Whh1, Wp_d1, G4H, Hdim);
  zero_state_kernel<<<(Bdim * Hdim) / 256, 256, 0, stream>>>(c0, c1, h0buf[0], h1buf[0]);

  dim3 cgrid(Bdim / 16, Hdim / 64);   // 16 x 8 blocks, 128 threads each

  int p = 0;
  // ---- encoder: 512 steps, two fused-cell kernels per step ----
  for (int t = 0; t < Tdim; ++t) {
    cell_l0_kernel<<<cgrid, 128, 0, stream>>>(Wp_e0, enc_b0, enc_Wih0,
                                              x + t, Tdim,
                                              h0buf[p], h0buf[p ^ 1], c0);
    cell_l1_kernel<<<cgrid, 128, 0, stream>>>(Wp_e1, enc_b1,
                                              h0buf[p ^ 1], h1buf[p], h1buf[p ^ 1], c1);
    p ^= 1;
  }
  // after 512 steps p == 0; encoder-final states live in buf[0] / c0 / c1

  // ---- decoder: 128 steps, cell0 + cell1 + linear head ----
  for (int s = 0; s < NPREDS; ++s) {
    const float* xp = (s == 0) ? (x + (Tdim - 1)) : prev;
    int xstride     = (s == 0) ? Tdim : 1;
    cell_l0_kernel<<<cgrid, 128, 0, stream>>>(Wp_d0, dec_b0, dec_Wih0,
                                              xp, xstride,
                                              h0buf[p], h0buf[p ^ 1], c0);
    cell_l1_kernel<<<cgrid, 128, 0, stream>>>(Wp_d1, dec_b1,
                                              h0buf[p ^ 1], h1buf[p], h1buf[p ^ 1], c1);
    pred_kernel<<<Bdim, 128, 0, stream>>>(h1buf[p ^ 1], lin_W, lin_b, out + s, prev);
    p ^= 1;
  }
}
